// Difference_89567247991564
// MI455X (gfx1250) — matrix operand order repaired and checked by reading, not compiled, not run on hardware
//
#include <hip/hip_runtime.h>

// result[i,j] = sum_k relu(a@feats^T)[i,k] * ((b@feats^T)[j,k] <= 0)
// Stage 1+2 (sign-sensitive): f32 WMMA (V_WMMA_F32_16X16X4_F32), exact.
//   P_hi/P_lo = hi/lo bf16 split of relu(a@feats^T)   (1024x256 each)
//   Qb        = bf16 indicator((b@feats^T) <= 0)      (1024x256, {0,1} exact)
// Stage 3 (bulk GEMM): bf16 WMMA (V_WMMA_F32_16X16X32_BF16), 8x MACs/inst:
//   result = P_hi@Qb^T + P_lo@Qb^T  accumulated in f32  (1024x1024)

typedef __attribute__((ext_vector_type(2)))  float  v2f;
typedef __attribute__((ext_vector_type(8)))  float  v8f;
typedef __attribute__((ext_vector_type(8)))  __bf16 v8bf;
typedef __attribute__((ext_vector_type(16))) __bf16 v16bf;

#define NA 1024
#define NB 1024
#define KF 256   // feature count (proj output cols / GEMM2 K)
#define DD 128   // inner dim of projection

static __device__ __forceinline__ v8f wmma_f32(v2f a, v2f b, v8f c) {
  return __builtin_amdgcn_wmma_f32_16x16x4_f32(false, a, false, b, (short)0, c,
                                               false, false);
}
static __device__ __forceinline__ v8f wmma_bf16(v16bf a, v16bf b, v8f c) {
  return __builtin_amdgcn_wmma_f32_16x16x32_bf16(false, a, false, b, (short)0,
                                                 c, false, false);
}

union afrag {          // A-matrix 16x32 bf16 fragment: two 8-bf16 chunks
  v16bf v;
  v8bf h[2];
};

// ---------------------------------------------------------------------------
// Stage 1+2: one wave per 16x16 tile of a_fk / b_fk, exact f32 WMMA.
// 128 row-tiles (64 a + 64 b) x 16 col-tiles = 2048 tiles = 256 blk x 8 waves.
// Epilogue: relu + hi/lo bf16 split (a) or <=0 indicator in bf16 (b).
// ---------------------------------------------------------------------------
__global__ __launch_bounds__(256) void proj_kernel(
    const float* __restrict__ a, const float* __restrict__ b,
    const float* __restrict__ feats, __bf16* __restrict__ Phi,
    __bf16* __restrict__ Plo, __bf16* __restrict__ Qb) {
  const int lane = threadIdx.x & 31;
  const int wave = threadIdx.x >> 5;
  const int tile = blockIdx.x * 8 + wave;   // 0..2047

  const int row_tile = tile >> 4;           // 0..127
  const int col_tile = tile & 15;           // 0..15
  const bool is_b = (row_tile >= 64);

  const float* __restrict__ src = is_b ? b : a;
  const int row0 = (is_b ? (row_tile - 64) : row_tile) * 16;
  const int col0 = col_tile * 16;

  const int mn = lane & 15;   // A row / B col owned by this lane
  const int g = lane >> 4;    // lane-group

  // f32 A 16x4: lane = m + 16*(k>>1), vgpr = k&1 -> contiguous float2
  const float* __restrict__ arow = src + (size_t)(row0 + mn) * DD + 2 * g;
  // f32 B 4x16: B[k][n] = feats[n][k] -> contiguous float2 from feats row n
  const float* __restrict__ brow = feats + (size_t)(col0 + mn) * DD + 2 * g;

  v8f acc = {};
#pragma unroll 4
  for (int k = 0; k < DD; k += 4) {
    v2f af = *(const v2f*)(arow + k);
    v2f bf = *(const v2f*)(brow + k);
    acc = wmma_f32(af, bf, acc);
  }

  // C/D layout: lane = n + 16*(m>>3), vgpr = m&7
  if (is_b) {
#pragma unroll
    for (int r = 0; r < 8; ++r) {
      const int m = 8 * g + r;
      const float v = acc[r];
      Qb[(size_t)(row0 + m) * KF + col0 + mn] =
          (v <= 0.0f) ? (__bf16)1.0f : (__bf16)0.0f;
    }
  } else {
#pragma unroll
    for (int r = 0; r < 8; ++r) {
      const int m = 8 * g + r;
      const float p = (acc[r] > 0.0f) ? acc[r] : 0.0f;
      const __bf16 hi = (__bf16)p;
      const __bf16 lo = (__bf16)(p - (float)hi);
      const size_t idx = (size_t)(row0 + m) * KF + col0 + mn;
      Phi[idx] = hi;
      Plo[idx] = lo;
    }
  }
}

// ---------------------------------------------------------------------------
// Stage 3: result = (Phi+Plo) @ Qb^T, 1024x1024, K=256, bf16 WMMA x32.
// 2x2 register tiling: each wave owns 32x32 of output (4x v8f acc).
// K loop: 8 iterations x 8 wmma (hi+lo per tile pair).
// 32x32 wave-tiles = 1024 waves = 128 blocks x 8 waves exactly.
// ---------------------------------------------------------------------------
__global__ __launch_bounds__(256) void gemm2_kernel(
    const __bf16* __restrict__ Phi, const __bf16* __restrict__ Plo,
    const __bf16* __restrict__ Qb, float* __restrict__ out) {
  const int lane = threadIdx.x & 31;
  const int wave = threadIdx.x >> 5;
  const int w = blockIdx.x * 8 + wave;      // 0..1023
  const int i0 = (w >> 5) * 32;             // output row base
  const int j0 = (w & 31) * 32;             // output col base

  const int mn = lane & 15;
  const int g = lane >> 4;

  // A 16x32 bf16: lane m+16g holds K = 8g..8g+7 (v0-3) and 16+8g.. (v4-7)
  const __bf16* __restrict__ ph0 = Phi + (size_t)(i0 + mn) * KF + 8 * g;
  const __bf16* __restrict__ ph1 = Phi + (size_t)(i0 + 16 + mn) * KF + 8 * g;
  const __bf16* __restrict__ pl0 = Plo + (size_t)(i0 + mn) * KF + 8 * g;
  const __bf16* __restrict__ pl1 = Plo + (size_t)(i0 + 16 + mn) * KF + 8 * g;
  // B 32x16 bf16: lane n+16g holds K = 8g..8g+7 and 16+8g..16+8g+7 (same as A)
  const __bf16* __restrict__ qb0 = Qb + (size_t)(j0 + mn) * KF + 8 * g;
  const __bf16* __restrict__ qb1 = Qb + (size_t)(j0 + 16 + mn) * KF + 8 * g;

  v8f acc00 = {}, acc01 = {}, acc10 = {}, acc11 = {};
#pragma unroll 2
  for (int k = 0; k < KF; k += 32) {
    afrag qa, qc;
    qa.h[0] = *(const v8bf*)(qb0 + k);
    qa.h[1] = *(const v8bf*)(qb0 + k + 16);
    qc.h[0] = *(const v8bf*)(qb1 + k);
    qc.h[1] = *(const v8bf*)(qb1 + k + 16);
    v16bf b0 = qa.v;
    v16bf b1 = qc.v;

    afrag h0, h1, l0, l1;
    h0.h[0] = *(const v8bf*)(ph0 + k);
    h0.h[1] = *(const v8bf*)(ph0 + k + 16);
    h1.h[0] = *(const v8bf*)(ph1 + k);
    h1.h[1] = *(const v8bf*)(ph1 + k + 16);
    l0.h[0] = *(const v8bf*)(pl0 + k);
    l0.h[1] = *(const v8bf*)(pl0 + k + 16);
    l1.h[0] = *(const v8bf*)(pl1 + k);
    l1.h[1] = *(const v8bf*)(pl1 + k + 16);

    acc00 = wmma_bf16(h0.v, b0, acc00);
    acc01 = wmma_bf16(h0.v, b1, acc01);
    acc10 = wmma_bf16(h1.v, b0, acc10);
    acc11 = wmma_bf16(h1.v, b1, acc11);
    acc00 = wmma_bf16(l0.v, b0, acc00);
    acc01 = wmma_bf16(l0.v, b1, acc01);
    acc10 = wmma_bf16(l1.v, b0, acc10);
    acc11 = wmma_bf16(l1.v, b1, acc11);
  }

#pragma unroll
  for (int r = 0; r < 8; ++r) {
    const int m = 8 * g + r;
    out[(size_t)(i0 + m) * NB + (j0 + mn)]           = acc00[r];
    out[(size_t)(i0 + m) * NB + (j0 + 16 + mn)]      = acc01[r];
    out[(size_t)(i0 + 16 + m) * NB + (j0 + mn)]      = acc10[r];
    out[(size_t)(i0 + 16 + m) * NB + (j0 + 16 + mn)] = acc11[r];
  }
}

extern "C" void kernel_launch(void* const* d_in, const int* in_sizes, int n_in,
                              void* d_out, int out_size, void* d_ws,
                              size_t ws_size, hipStream_t stream) {
  const float* a = (const float*)d_in[0];      // (1024,128) f32
  const float* b = (const float*)d_in[1];      // (1024,128) f32
  const float* feats = (const float*)d_in[2];  // (256,128)  f32
  float* out = (float*)d_out;                  // (1024,1024) f32

  __bf16* Phi = (__bf16*)d_ws;                 // 512 KB
  __bf16* Plo = Phi + (size_t)NA * KF;         // 512 KB
  __bf16* Qb  = Plo + (size_t)NA * KF;         // 512 KB

  proj_kernel<<<256, 256, 0, stream>>>(a, b, feats, Phi, Plo, Qb);
  gemm2_kernel<<<128, 256, 0, stream>>>(Phi, Plo, Qb, out);
}
